// GNNModel_23373212025374
// MI455X (gfx1250) — compile-verified, matrix operand order checked by reading, not run on hardware
//
#include <hip/hip_runtime.h>
#include <hip/hip_bf16.h>
#include <stdint.h>

#define N_NODES  20000
#define N_EDGES  320000
#define F_INDIM  64
#define HDIM     512
#define G_GRAPHS 512
#define BN_EPS   1e-5f

typedef __attribute__((ext_vector_type(16))) __bf16 v16bf;
typedef __attribute__((ext_vector_type(8)))  float  v8f;

__device__ __forceinline__ uint16_t f2bf(float f) {
    uint32_t u = __float_as_uint(f);
    u += 0x7FFFu + ((u >> 16) & 1u);   // round-to-nearest-even
    return (uint16_t)(u >> 16);
}

// ---------------------------------------------------------------- utilities
__global__ void zero_i32_kernel(int* __restrict__ p, int n) {
    int i = blockIdx.x * blockDim.x + threadIdx.x;
    if (i < n) p[i] = 0;
}

__global__ void copy_i32_kernel(const int* __restrict__ in, int* __restrict__ out, int n) {
    int i = blockIdx.x * blockDim.x + threadIdx.x;
    if (i < n) out[i] = in[i];
}

__global__ void count_i32_kernel(const int* __restrict__ idx, int* __restrict__ cnt, int n) {
    int i = blockIdx.x * blockDim.x + threadIdx.x;
    if (i < n) atomicAdd(&cnt[idx[i]], 1);       // int atomics: deterministic
}

__global__ void dinv_kernel(const int* __restrict__ deg, float* __restrict__ dinv, int n) {
    int i = blockIdx.x * blockDim.x + threadIdx.x;
    if (i < n) dinv[i] = rsqrtf((float)deg[i] + 1.0f);   // self-loop: deg+1
}

__global__ void f32_to_bf16_kernel(const float* __restrict__ in, uint16_t* __restrict__ out, long n) {
    long i = (long)blockIdx.x * blockDim.x + threadIdx.x;
    if (i < n) out[i] = f2bf(in[i]);
}

// W[k, c] (K x NO row-major) -> Wt[c, k] (NO x K row-major, bf16)
__global__ void transpose_to_bf16_kernel(const float* __restrict__ W, uint16_t* __restrict__ Wt,
                                         int K, int NO) {
    int i = blockIdx.x * blockDim.x + threadIdx.x;
    if (i >= K * NO) return;
    int k = i % K;
    int c = i / K;
    Wt[i] = f2bf(W[(long)k * NO + c]);
}

// Single-block exclusive scan (int), writes out[0..n-1] and total at out[n].
__global__ void exclusive_scan_kernel(const int* __restrict__ in, int* __restrict__ out, int n) {
    __shared__ int wsum[32];
    __shared__ int carry;
    const int tid  = threadIdx.x;            // block = 1024
    const int lane = tid & 31;
    const int wid  = tid >> 5;
    const int nw   = blockDim.x >> 5;
    if (tid == 0) carry = 0;
    __syncthreads();
    for (int base = 0; base < n; base += blockDim.x) {
        int i = base + tid;
        int v = (i < n) ? in[i] : 0;
        int s = v;                            // inclusive wave scan
        for (int o = 1; o < 32; o <<= 1) {
            int t = __shfl_up(s, o, 32);
            if (lane >= o) s += t;
        }
        if (lane == 31) wsum[wid] = s;
        __syncthreads();
        if (wid == 0) {
            int ws = (lane < nw) ? wsum[lane] : 0;
            for (int o = 1; o < 32; o <<= 1) {
                int t = __shfl_up(ws, o, 32);
                if (lane >= o) ws += t;
            }
            if (lane < nw) wsum[lane] = ws;   // inclusive across waves
        }
        __syncthreads();
        int waveoff = (wid > 0) ? wsum[wid - 1] : 0;
        if (i < n) out[i] = carry + waveoff + s - v;
        __syncthreads();
        if (tid == 0) carry += wsum[nw - 1];
        __syncthreads();
    }
    if (tid == 0) out[n] = carry;
}

// eidx[rowstart[dst[e]] + k] = e  (CSR by destination; built once per launch)
__global__ void fill_csr_kernel(const int* __restrict__ dst, int* __restrict__ cursor,
                                int* __restrict__ eidx, int E) {
    int e = blockIdx.x * blockDim.x + threadIdx.x;
    if (e < E) {
        int p = atomicAdd(&cursor[dst[e]], 1);
        eidx[p] = e;
    }
}

// ---------------------------------------------------------------- WMMA GEMM
// C[M,NO] (f32) = A[M,K] (bf16 row-major) @ Bt[NO,K]^T (bf16, W pre-transposed).
// One wave computes a 32x32 block: 2x2 register tiles, 4 WMMAs / 32-K step.
__global__ void gemm_wmma_bf16_kernel(const uint16_t* __restrict__ A,
                                      const uint16_t* __restrict__ Bt,
                                      float* __restrict__ C,
                                      int M, int K, int NO) {
    int gwave  = (int)(((long)blockIdx.x * blockDim.x + threadIdx.x) >> 5);
    int blksN  = NO >> 5;
    int total  = (M >> 5) * blksN;
    if (gwave >= total) return;               // whole-wave exit: EXEC stays all-ones
    int lane = threadIdx.x & 31;
    int bm   = gwave / blksN;
    int bn   = gwave - bm * blksN;
    int half = lane >> 4;                     // K-half select per ISA 16-bit A layout
    int r    = lane & 15;

    const uint16_t* a0 = A  + (long)(bm * 32 + r) * K + half * 8;
    const uint16_t* a1 = a0 + (long)16 * K;
    const uint16_t* b0 = Bt + (long)(bn * 32 + r) * K + half * 8;
    const uint16_t* b1 = b0 + (long)16 * K;

    v8f c00 = {0.f,0.f,0.f,0.f,0.f,0.f,0.f,0.f};
    v8f c01 = c00, c10 = c00, c11 = c00;

    for (int kb = 0; kb < K; kb += 32) {
        union { uint4 q[2]; v16bf v; } ua0, ua1, ub0, ub1;
        ua0.q[0] = *(const uint4*)(a0 + kb);
        ua0.q[1] = *(const uint4*)(a0 + kb + 16);
        ua1.q[0] = *(const uint4*)(a1 + kb);
        ua1.q[1] = *(const uint4*)(a1 + kb + 16);
        ub0.q[0] = *(const uint4*)(b0 + kb);
        ub0.q[1] = *(const uint4*)(b0 + kb + 16);
        ub1.q[0] = *(const uint4*)(b1 + kb);
        ub1.q[1] = *(const uint4*)(b1 + kb + 16);
        c00 = __builtin_amdgcn_wmma_f32_16x16x32_bf16(false, ua0.v, false, ub0.v, (short)0, c00, false, false);
        c01 = __builtin_amdgcn_wmma_f32_16x16x32_bf16(false, ua0.v, false, ub1.v, (short)0, c01, false, false);
        c10 = __builtin_amdgcn_wmma_f32_16x16x32_bf16(false, ua1.v, false, ub0.v, (short)0, c10, false, false);
        c11 = __builtin_amdgcn_wmma_f32_16x16x32_bf16(false, ua1.v, false, ub1.v, (short)0, c11, false, false);
    }

    // C tile layout: VGPR i -> row (row0 + half*8 + i), col (col0 + (lane&15))
    int R0 = bm * 32, C0 = bn * 32;
#define STORE_TILE(acc, row0, col0)                                        \
    {                                                                      \
        int col = (col0) + r;                                              \
        int rw  = (row0) + half * 8;                                       \
        _Pragma("unroll")                                                  \
        for (int i = 0; i < 8; ++i)                                        \
            C[(long)(rw + i) * NO + col] = (acc)[i];                       \
    }
    STORE_TILE(c00, R0,      C0)
    STORE_TILE(c01, R0,      C0 + 16)
    STORE_TILE(c10, R0 + 16, C0)
    STORE_TILE(c11, R0 + 16, C0 + 16)
#undef STORE_TILE
}

// ---------------------------------------------------------------- fused GCN aggregation
// For node n, channels [c4..c4+3]:
//   s   = sum_{e in CSR row n} hw[src[e]] * dinv[src]*dinv[n]     (gather, no atomics)
//   val = relu(BN(s + hw[n]*dinv[n]^2 + bias))
// writes fp32 (pooling) + bf16 (next GEMM input)
__global__ void gather_bn_relu_kernel(const int* __restrict__ rowstart, const int* __restrict__ eidx,
                                      const int* __restrict__ src, const float* __restrict__ dinv,
                                      const float* __restrict__ hw,
                                      const float* __restrict__ bias,
                                      const float* __restrict__ g,  const float* __restrict__ be,
                                      const float* __restrict__ m,  const float* __restrict__ vv,
                                      float* __restrict__ hfeat, uint16_t* __restrict__ hbf,
                                      int Nn, int H) {
    long gid = (long)blockIdx.x * blockDim.x + threadIdx.x;
    int  cpe = H >> 2;
    if (gid >= (long)Nn * cpe) return;
    int n  = (int)(gid / cpe);
    int c4 = ((int)(gid - (long)n * cpe)) << 2;

    float dn  = dinv[n];
    int beg = rowstart[n], end = rowstart[n + 1];
    float4 s = {0.f, 0.f, 0.f, 0.f};
    for (int p = beg; p < end; ++p) {
        int e  = eidx[p];
        int sn = src[e];
        float en = dinv[sn] * dn;
        float4 v = *(const float4*)(hw + (long)sn * H + c4);
        s.x += v.x * en; s.y += v.y * en; s.z += v.z * en; s.w += v.w * en;
    }
    float  snorm = dn * dn;
    float4 hv = *(const float4*)(hw + (long)n * H + c4);
    float4 bi = *(const float4*)(bias + c4);
    float4 gg = *(const float4*)(g + c4);
    float4 bb = *(const float4*)(be + c4);
    float4 mm = *(const float4*)(m + c4);
    float4 va = *(const float4*)(vv + c4);
    float4 o;
    o.x = fmaxf((s.x + hv.x * snorm + bi.x - mm.x) * (gg.x * rsqrtf(va.x + BN_EPS)) + bb.x, 0.f);
    o.y = fmaxf((s.y + hv.y * snorm + bi.y - mm.y) * (gg.y * rsqrtf(va.y + BN_EPS)) + bb.y, 0.f);
    o.z = fmaxf((s.z + hv.z * snorm + bi.z - mm.z) * (gg.z * rsqrtf(va.z + BN_EPS)) + bb.z, 0.f);
    o.w = fmaxf((s.w + hv.w * snorm + bi.w - mm.w) * (gg.w * rsqrtf(va.w + BN_EPS)) + bb.w, 0.f);
    *(float4*)(hfeat + (long)n * H + c4) = o;
    uint16_t* hb = hbf + (long)n * H + c4;
    hb[0] = f2bf(o.x); hb[1] = f2bf(o.y); hb[2] = f2bf(o.z); hb[3] = f2bf(o.w);
}

// ---------------------------------------------------------------- pooling (batch is sorted)
__global__ void pool_mean_kernel(const float* __restrict__ h, const int* __restrict__ gstart,
                                 uint16_t* __restrict__ pbf, int H) {
    int g = blockIdx.x;
    int beg = gstart[g], end = gstart[g + 1];
    float inv = 1.0f / fmaxf((float)(end - beg), 1.0f);
    for (int c = threadIdx.x; c < H; c += blockDim.x) {
        float s = 0.0f;
        for (int n = beg; n < end; ++n) s += h[(long)n * H + c];
        pbf[(long)g * H + c] = f2bf(s * inv);
    }
}

// out[g] = relu(mlp1[g,:] + lin1_b) . lin2_w + lin2_b   (one wave per graph)
__global__ void head_kernel(const float* __restrict__ mlp1, const float* __restrict__ b1,
                            const float* __restrict__ w2, const float* __restrict__ b2,
                            float* __restrict__ out, int Gg, int Hh) {
    int wave = (int)(((long)blockIdx.x * blockDim.x + threadIdx.x) >> 5);
    int lane = threadIdx.x & 31;
    if (wave >= Gg) return;
    float s = 0.0f;
    for (int c = lane; c < Hh; c += 32) {
        float v = fmaxf(mlp1[(long)wave * Hh + c] + b1[c], 0.0f);
        s += v * w2[c];
    }
    for (int o = 16; o > 0; o >>= 1) s += __shfl_xor(s, o, 32);
    if (lane == 0) out[wave] = s + b2[0];
}

// ---------------------------------------------------------------- launch
extern "C" void kernel_launch(void* const* d_in, const int* in_sizes, int n_in,
                              void* d_out, int out_size, void* d_ws, size_t ws_size,
                              hipStream_t stream) {
    (void)in_sizes; (void)n_in; (void)out_size; (void)ws_size;
    const float* x     = (const float*)d_in[0];
    const int*   ei    = (const int*)d_in[1];
    const int*   batch = (const int*)d_in[2];
    const float* W0 = (const float*)d_in[3];  const float* b0 = (const float*)d_in[4];
    const float* W1 = (const float*)d_in[5];  const float* b1 = (const float*)d_in[6];
    const float* W2 = (const float*)d_in[7];  const float* b2 = (const float*)d_in[8];
    const float* g0 = (const float*)d_in[9];  const float* be0 = (const float*)d_in[10];
    const float* m0 = (const float*)d_in[11]; const float* v0  = (const float*)d_in[12];
    const float* g1 = (const float*)d_in[13]; const float* be1 = (const float*)d_in[14];
    const float* m1 = (const float*)d_in[15]; const float* v1  = (const float*)d_in[16];
    const float* g2 = (const float*)d_in[17]; const float* be2 = (const float*)d_in[18];
    const float* m2 = (const float*)d_in[19]; const float* v2  = (const float*)d_in[20];
    const float* lin1w = (const float*)d_in[21]; const float* lin1b = (const float*)d_in[22];
    const float* lin2w = (const float*)d_in[23]; const float* lin2b = (const float*)d_in[24];

    char* wsp = (char*)d_ws;
    auto alloc = [&](size_t bytes) -> char* {
        char* p = wsp; wsp += (bytes + 255) & ~(size_t)255; return p;
    };
    int*      deg_i    = (int*)     alloc((size_t)N_NODES * 4);
    int*      rowstart = (int*)     alloc((size_t)(N_NODES + 1) * 4);
    int*      cursor   = (int*)     alloc((size_t)N_NODES * 4);
    int*      eidx     = (int*)     alloc((size_t)N_EDGES * 4);
    float*    dinv     = (float*)   alloc((size_t)N_NODES * 4);
    uint16_t* xbf      = (uint16_t*)alloc((size_t)N_NODES * F_INDIM * 2);
    uint16_t* hbf      = (uint16_t*)alloc((size_t)N_NODES * HDIM * 2);
    float*    hw       = (float*)   alloc((size_t)N_NODES * HDIM * 4);   // GEMM output
    float*    hfeat    = (float*)   alloc((size_t)N_NODES * HDIM * 4);   // post-activation
    uint16_t* W0t      = (uint16_t*)alloc((size_t)F_INDIM * HDIM * 2);
    uint16_t* W1t      = (uint16_t*)alloc((size_t)HDIM * HDIM * 2);
    uint16_t* W2t      = (uint16_t*)alloc((size_t)HDIM * HDIM * 2);
    uint16_t* L1t      = (uint16_t*)alloc((size_t)HDIM * (HDIM / 2) * 2);
    int*      gcnt     = (int*)     alloc((size_t)G_GRAPHS * 4);
    int*      gstart   = (int*)     alloc((size_t)(G_GRAPHS + 1) * 4);
    uint16_t* pbf      = (uint16_t*)alloc((size_t)G_GRAPHS * HDIM * 2);
    float*    mlp1     = (float*)   alloc((size_t)G_GRAPHS * (HDIM / 2) * 4);

    const int* srcv = ei;
    const int* dstv = ei + N_EDGES;
    const int B = 256;

    // ---- degree, normalization, CSR-by-dst (once; dst fixed for all layers)
    zero_i32_kernel<<<(N_NODES + B - 1) / B, B, 0, stream>>>(deg_i, N_NODES);
    count_i32_kernel<<<(N_EDGES + B - 1) / B, B, 0, stream>>>(dstv, deg_i, N_EDGES);
    dinv_kernel<<<(N_NODES + B - 1) / B, B, 0, stream>>>(deg_i, dinv, N_NODES);
    exclusive_scan_kernel<<<1, 1024, 0, stream>>>(deg_i, rowstart, N_NODES);
    copy_i32_kernel<<<(N_NODES + B - 1) / B, B, 0, stream>>>(rowstart, cursor, N_NODES);
    fill_csr_kernel<<<(N_EDGES + B - 1) / B, B, 0, stream>>>(dstv, cursor, eidx, N_EDGES);

    // ---- graph ranges for pooling (batch is sorted)
    zero_i32_kernel<<<(G_GRAPHS + B - 1) / B, B, 0, stream>>>(gcnt, G_GRAPHS);
    count_i32_kernel<<<(N_NODES + B - 1) / B, B, 0, stream>>>(batch, gcnt, N_NODES);
    exclusive_scan_kernel<<<1, 1024, 0, stream>>>(gcnt, gstart, G_GRAPHS);

    // ---- bf16 conversions / weight transposes
    long nx = (long)N_NODES * F_INDIM;
    f32_to_bf16_kernel<<<(unsigned)((nx + B - 1) / B), B, 0, stream>>>(x, xbf, nx);
    transpose_to_bf16_kernel<<<(F_INDIM * HDIM + B - 1) / B, B, 0, stream>>>(W0, W0t, F_INDIM, HDIM);
    transpose_to_bf16_kernel<<<(HDIM * HDIM + B - 1) / B, B, 0, stream>>>(W1, W1t, HDIM, HDIM);
    transpose_to_bf16_kernel<<<(HDIM * HDIM + B - 1) / B, B, 0, stream>>>(W2, W2t, HDIM, HDIM);
    transpose_to_bf16_kernel<<<(HDIM * (HDIM / 2) + B - 1) / B, B, 0, stream>>>(lin1w, L1t, HDIM, HDIM / 2);

    // ---- 3 GCN layers: WMMA GEMM -> fused gather+BN+ReLU (no float atomics)
    auto run_layer = [&](const uint16_t* Ain, int K,
                         const uint16_t* Wt, const float* bias,
                         const float* g, const float* be, const float* m, const float* v) {
        long waves = (long)(N_NODES / 32) * (HDIM / 32);
        gemm_wmma_bf16_kernel<<<(unsigned)((waves * 32 + B - 1) / B), B, 0, stream>>>(
            Ain, Wt, hw, N_NODES, K, HDIM);
        long gthr = (long)N_NODES * (HDIM / 4);
        gather_bn_relu_kernel<<<(unsigned)((gthr + B - 1) / B), B, 0, stream>>>(
            rowstart, eidx, srcv, dinv, hw, bias, g, be, m, v, hfeat, hbf, N_NODES, HDIM);
    };
    run_layer(xbf, F_INDIM, W0t, b0, g0, be0, m0, v0);
    run_layer(hbf, HDIM,    W1t, b1, g1, be1, m1, v1);
    run_layer(hbf, HDIM,    W2t, b2, g2, be2, m2, v2);

    // ---- mean pool (deterministic, contiguous graph ranges)
    pool_mean_kernel<<<G_GRAPHS, 256, 0, stream>>>(hfeat, gstart, pbf, HDIM);

    // ---- MLP head: [G,512] @ [512,256] via WMMA, then fused relu-dot
    long waves2 = (long)(G_GRAPHS / 32) * ((HDIM / 2) / 32);
    gemm_wmma_bf16_kernel<<<(unsigned)((waves2 * 32 + B - 1) / B), B, 0, stream>>>(
        pbf, L1t, mlp1, G_GRAPHS, HDIM, HDIM / 2);
    head_kernel<<<(G_GRAPHS * 32 + B - 1) / B, B, 0, stream>>>(
        mlp1, lin1b, lin2w, lin2b, (float*)d_out, G_GRAPHS, HDIM / 2);
}
